// Gabor_LSM_partition_12627203850877
// MI455X (gfx1250) — matrix operand order repaired and checked by reading, not compiled, not run on hardware
//
#include <hip/hip_runtime.h>

typedef __attribute__((ext_vector_type(16))) _Float16 v16h;
typedef __attribute__((ext_vector_type(8)))  _Float16 v8h;
typedef __attribute__((ext_vector_type(8)))  float    v8f;

// ---- static config (matches reference) ----
#define T_STEPS 48
#define BATCH   64
#define NRES    1000
#define N_PAD   1024       // NRES padded to multiple of 128
#define NPART   4
#define OCH     16
#define HH      64
#define WW      64
#define KDIM    7
#define HO      29
#define WO      29
#define IN_SZ   13456      // OCH*HO*WO
#define K_PAD   13472      // IN_SZ rounded up to multiple of 32
#define ALPHA   0.9f
#define BETA    0.9f
#define THRESH  1.0f

#define USE_ASYNC 1

// ---- global -> LDS staging helper (16 bytes per call, per lane) ----
__device__ __forceinline__ void stage16(const _Float16* __restrict__ g, _Float16* l) {
#if USE_ASYNC
    // CDNA5 async copy: LDS byte address in vdst VGPR (low 32 bits of generic
    // address == LDS offset per aperture rule), 64-bit global vaddr, no saddr.
    asm volatile("global_load_async_to_lds_b128 %0, %1, off"
                 :: "v"((unsigned)(uintptr_t)l),
                    "v"((unsigned long long)(uintptr_t)g)
                 : "memory");
#else
    *(v8h*)l = *(const v8h*)g;
#endif
}
__device__ __forceinline__ void stage_wait() {
#if USE_ASYNC
    asm volatile("s_wait_asynccnt 0x0" ::: "memory");
#endif
}

// ---------------- pad-convert Wins: [P][NRES][IN_SZ] f32 -> [P][N_PAD][K_PAD] f16 ----
__global__ void pad_cvt_wins(const float* __restrict__ src, _Float16* __restrict__ dst) {
    const int k = blockIdx.x * blockDim.x + threadIdx.x;
    const int n = blockIdx.y;
    const int p = blockIdx.z;
    if (k >= K_PAD) return;
    float v = 0.f;
    if (n < NRES && k < IN_SZ) v = src[((size_t)p * NRES + n) * IN_SZ + k];
    dst[((size_t)p * N_PAD + n) * K_PAD + k] = (_Float16)v;
}

// ---------------- pad-convert Wlsm: [NRES][NRES] f32 -> [N_PAD][N_PAD] f16 ----
__global__ void pad_cvt_wlsm(const float* __restrict__ src, _Float16* __restrict__ dst) {
    const int k = blockIdx.x * blockDim.x + threadIdx.x;
    const int n = blockIdx.y;
    float v = 0.f;
    if (n < NRES && k < NRES) v = src[(size_t)n * NRES + k];
    dst[(size_t)n * N_PAD + k] = (_Float16)v;
}

// ---------------- zero state + A K-pad region + padded f16 spike buffers ----------------
__global__ void init_state(float* __restrict__ syn, float* __restrict__ mem,
                           _Float16* __restrict__ spk0h, _Float16* __restrict__ spk1h,
                           _Float16* __restrict__ A) {
    int i = blockIdx.x * blockDim.x + threadIdx.x;
    if (i < BATCH * NRES) { syn[i] = 0.f; mem[i] = 0.f; }
    if (i < BATCH * N_PAD) { spk0h[i] = (_Float16)0.f; spk1h[i] = (_Float16)0.f; }
    const int npad = T_STEPS * BATCH * (K_PAD - IN_SZ); // 49152
    if (i < npad) {
        int row = i >> 4;
        int k   = IN_SZ + (i & 15);
        A[(size_t)row * K_PAD + k] = (_Float16)0.f;
    }
}

// ---------------- Gabor conv: one block per (t,b) image ----------------
// Writes f16 directly into GEMM A layout: A[tb][oc*841 + ho*29 + wo]
__global__ void gabor_conv(const float* __restrict__ x, const float* __restrict__ w,
                           _Float16* __restrict__ A) {
    __shared__ float xs[HH * WW];
    __shared__ float wf[OCH * KDIM * KDIM];
    const int tb = blockIdx.x;
    const float* xp = x + (size_t)tb * HH * WW;
    for (int i = threadIdx.x; i < HH * WW; i += blockDim.x) xs[i] = xp[i];
    for (int i = threadIdx.x; i < OCH * KDIM * KDIM; i += blockDim.x) wf[i] = w[i];
    __syncthreads();
    for (int pos = threadIdx.x; pos < HO * WO; pos += blockDim.x) {
        int ho = pos / WO, wo = pos - ho * WO;
        int ih = ho * 2, iw = wo * 2;
        float acc[OCH];
#pragma unroll
        for (int o = 0; o < OCH; o++) acc[o] = 0.f;
        for (int kh = 0; kh < KDIM; kh++) {
#pragma unroll
            for (int kw = 0; kw < KDIM; kw++) {
                float xv = xs[(ih + kh) * WW + iw + kw];
#pragma unroll
                for (int o = 0; o < OCH; o++)
                    acc[o] = fmaf(xv, wf[o * 49 + kh * 7 + kw], acc[o]);
            }
        }
        _Float16* ap = A + (size_t)tb * K_PAD + pos;
#pragma unroll
        for (int o = 0; o < OCH; o++) ap[o * (HO * WO)] = (_Float16)acc[o];
    }
}

// ---------------- Projection GEMM: curr = feat @ Wins[p]^T + fc1_b ----------------
// Per partition p: C[768 x NRES] = A[768 x K_PAD f16] x WinsPad[p][N_PAD x K_PAD]^T
// Block: 128 threads (4 waves). Block tile M=128, N=64; wave tile 32x64 (8 WMMA/K-step).
// Double-buffered LDS: async loads of tile i+1 overlap WMMAs on tile i.
__global__ void __launch_bounds__(128) proj_gemm(const _Float16* __restrict__ A,
                                                 const _Float16* __restrict__ Bm,
                                                 const float* __restrict__ bias,
                                                 float* __restrict__ curr) {
    __shared__ _Float16 As[2][128][40];
    __shared__ _Float16 Bs[2][64][40];
    const int tid  = threadIdx.x;
    const int wave = tid >> 5;
    const int lane = tid & 31;
    const int n0 = blockIdx.x * 64;
    const int m0 = blockIdx.y * 128;
    const int p  = blockIdx.z;
    const size_t arow0 = (size_t)p * 768 + m0;   // global row base ( == t*64+b )
    const size_t brow0 = (size_t)p * N_PAD + n0;
    const int halfsel = lane >> 4;
    const int l15 = lane & 15;

    // per-thread staging coordinates (chunk c covers rows c*32 .. c*32+31)
    const int srow = tid >> 2;          // 0..31
    const int soff = (tid & 3) * 8;     // 0,8,16,24

    auto issue = [&](int buf, int kk) {
#pragma unroll
        for (int c = 0; c < 4; c++)     // A: 128 rows
            stage16(A + (arow0 + c * 32 + srow) * K_PAD + kk + soff,
                    &As[buf][c * 32 + srow][soff]);
#pragma unroll
        for (int c = 0; c < 2; c++)     // B: 64 rows
            stage16(Bm + (brow0 + c * 32 + srow) * K_PAD + kk + soff,
                    &Bs[buf][c * 32 + srow][soff]);
    };

    v8f acc[2][4] = {};

    issue(0, 0);
    int buf = 0;
    for (int kk = 0; kk < K_PAD; kk += 32, buf ^= 1) {
        stage_wait();        // this wave's async loads for tile `buf` landed
        __syncthreads();     // all waves' loads visible; prior reads of buf^1 done
        if (kk + 32 < K_PAD) issue(buf ^ 1, kk + 32);

        // A fragments: lanes 0-15 K {0..15}; lanes 16-31 K {16..31}
        const int klo = halfsel * 16;
        v16h af[2];
#pragma unroll
        for (int h = 0; h < 2; h++) {
            int arow = wave * 32 + h * 16 + l15;
#pragma unroll
            for (int e = 0; e < 8; e++) {
                af[h][e]     = As[buf][arow][klo + e];
                af[h][e + 8] = As[buf][arow][klo + 8 + e];
            }
        }
#pragma unroll
        for (int nt = 0; nt < 4; nt++) {
            const int brow = nt * 16 + l15;
            const int kb   = halfsel * 16;
            v16h bf;
#pragma unroll
            for (int e = 0; e < 16; e++) bf[e] = Bs[buf][brow][kb + e];
#pragma unroll
            for (int h = 0; h < 2; h++)
                acc[h][nt] = __builtin_amdgcn_wmma_f32_16x16x32_f16(
                    false, af[h], false, bf, (short)0, acc[h][nt], false, false);
        }
        __syncthreads();     // done reading tile `buf` before it is refilled
    }
    // epilogue: f32 C layout: VGPR v -> M = v (+8 for lanes 16-31), N = lane&15
#pragma unroll
    for (int nt = 0; nt < 4; nt++) {
        int col = n0 + nt * 16 + l15;
        if (col < NRES) {
            float b = bias[col];
#pragma unroll
            for (int h = 0; h < 2; h++) {
#pragma unroll
                for (int v = 0; v < 8; v++) {
                    int mrow = wave * 32 + h * 16 + v + halfsel * 8;
                    curr[(arow0 + mrow) * NRES + col] = acc[h][nt][v] + b;
                }
            }
        }
    }
}

// ---------------- One recurrent step: rec = spk @ Wlsm^T, fused neuron update ----------------
// M=64 (batch), K=N_PAD, N=N_PAD. 8 blocks x 128 threads; wave tile 16x128 (8 WMMA/K-step).
// Spikes kept as padded f16 [64][N_PAD] ping/pong (pad region stays zero).
__global__ void __launch_bounds__(128) rsyn_step(const float* __restrict__ curr,
                                                 const _Float16* __restrict__ Wl,
                                                 const float* __restrict__ rb,
                                                 float* __restrict__ syn,
                                                 float* __restrict__ mem,
                                                 const _Float16* __restrict__ spk_in,
                                                 _Float16* __restrict__ spk_out,
                                                 float* __restrict__ out, int t) {
    __shared__ _Float16 Ss[2][64][40];
    __shared__ _Float16 Ws[2][128][40];
    const int tid  = threadIdx.x;
    const int wave = tid >> 5;
    const int lane = tid & 31;
    const int n0 = blockIdx.x * 128;
    const int halfsel = lane >> 4;
    const int l15 = lane & 15;

    const int srow = tid >> 2;
    const int soff = (tid & 3) * 8;

    auto issue = [&](int buf, int kk) {
#pragma unroll
        for (int c = 0; c < 2; c++)     // spikes: 64 rows
            stage16(spk_in + (size_t)(c * 32 + srow) * N_PAD + kk + soff,
                    &Ss[buf][c * 32 + srow][soff]);
#pragma unroll
        for (int c = 0; c < 4; c++)     // Wlsm: 128 rows
            stage16(Wl + (size_t)(n0 + c * 32 + srow) * N_PAD + kk + soff,
                    &Ws[buf][c * 32 + srow][soff]);
    };

    v8f acc[8] = {};

    issue(0, 0);
    int buf = 0;
    for (int kk = 0; kk < N_PAD; kk += 32, buf ^= 1) {
        stage_wait();
        __syncthreads();
        if (kk + 32 < N_PAD) issue(buf ^ 1, kk + 32);

        const int arow = wave * 16 + l15;
        const int klo  = halfsel * 16;
        v16h af;
#pragma unroll
        for (int e = 0; e < 8; e++) {
            af[e]     = Ss[buf][arow][klo + e];
            af[e + 8] = Ss[buf][arow][klo + 8 + e];
        }
#pragma unroll
        for (int nt = 0; nt < 8; nt++) {
            const int brow = nt * 16 + l15;
            const int kb   = halfsel * 16;
            v16h bf;
#pragma unroll
            for (int e = 0; e < 16; e++) bf[e] = Ws[buf][brow][kb + e];
            acc[nt] = __builtin_amdgcn_wmma_f32_16x16x32_f16(
                false, af, false, bf, (short)0, acc[nt], false, false);
        }
        __syncthreads();
    }

    // fused RSynaptic update (reset uses mem_prev; spk via Heaviside)
    const float* currt = curr + (size_t)t * BATCH * NRES;
    float* outt = out + (size_t)t * BATCH * NRES;
#pragma unroll
    for (int nt = 0; nt < 8; nt++) {
        int n = n0 + nt * 16 + l15;
        if (n < NRES) {
            float rbn = rb[n];
#pragma unroll
            for (int v = 0; v < 8; v++) {
                int b = wave * 16 + v + halfsel * 8;
                size_t i = (size_t)b * NRES + n;
                float s = ALPHA * syn[i] + currt[i] + acc[nt][v] + rbn;
                float mo = mem[i];
                float reset = (mo - THRESH > 0.f) ? 1.f : 0.f;
                float m = BETA * mo + s - reset * THRESH;
                float sp = (m - THRESH > 0.f) ? 1.f : 0.f;
                syn[i] = s;
                mem[i] = m;
                spk_out[(size_t)b * N_PAD + n] = (_Float16)sp;
                outt[i] = sp;
            }
        }
    }
}

extern "C" void kernel_launch(void* const* d_in, const int* in_sizes, int n_in,
                              void* d_out, int out_size, void* d_ws, size_t ws_size,
                              hipStream_t stream) {
    const float* x    = (const float*)d_in[0];
    const float* gw   = (const float*)d_in[1];
    const float* Wins = (const float*)d_in[2];
    const float* fc1b = (const float*)d_in[3];
    const float* Wlsm = (const float*)d_in[4];
    const float* recb = (const float*)d_in[5];
    float* out = (float*)d_out;

    char* ws = (char*)d_ws;
    size_t off = 0;
    auto alloc = [&](size_t bytes) -> void* {
        void* p = ws + off;
        off = (off + bytes + 255) & ~(size_t)255;
        return p;
    };
    _Float16* A    = (_Float16*)alloc((size_t)T_STEPS * BATCH * K_PAD * 2);   // 82.8 MB
    _Float16* Bm   = (_Float16*)alloc((size_t)NPART * N_PAD * K_PAD * 2);     // 110.4 MB
    _Float16* Wl   = (_Float16*)alloc((size_t)N_PAD * N_PAD * 2);             // 2 MB
    float* curr    = (float*)alloc((size_t)T_STEPS * BATCH * NRES * 4);       // 12.3 MB
    float* syn     = (float*)alloc((size_t)BATCH * NRES * 4);
    float* mem     = (float*)alloc((size_t)BATCH * NRES * 4);
    _Float16* spk0 = (_Float16*)alloc((size_t)BATCH * N_PAD * 2);
    _Float16* spk1 = (_Float16*)alloc((size_t)BATCH * N_PAD * 2);

    pad_cvt_wins<<<dim3((K_PAD + 255) / 256, N_PAD, NPART), 256, 0, stream>>>(Wins, Bm);
    pad_cvt_wlsm<<<dim3(N_PAD / 256, N_PAD), 256, 0, stream>>>(Wlsm, Wl);

    gabor_conv<<<T_STEPS * BATCH, 256, 0, stream>>>(x, gw, A);
    init_state<<<(BATCH * N_PAD * 2 + 255) / 256, 256, 0, stream>>>(syn, mem, spk0, spk1, A);

    proj_gemm<<<dim3(16, 6, NPART), 128, 0, stream>>>(A, Bm, fc1b, curr);

    for (int t = 0; t < T_STEPS; t++) {
        const _Float16* si = (t & 1) ? spk1 : spk0;
        _Float16*       so = (t & 1) ? spk0 : spk1;
        rsyn_step<<<8, 128, 0, stream>>>(curr, Wl, recb, syn, mem, si, so, out, t);
    }
}
